// AdaptiveCausalAttention_2018634629580
// MI455X (gfx1250) — matrix operand order repaired and checked by reading, not compiled, not run on hardware
//
#include <hip/hip_runtime.h>
#include <math.h>

typedef __attribute__((ext_vector_type(16))) __bf16 v16bf;
typedef __attribute__((ext_vector_type(8)))  __bf16 v8bf;
typedef __attribute__((ext_vector_type(8)))  float  v8f;

#define N_HEAD  16
#define T_SEQ   2048
#define C_EMB   1024
#define D_HEAD  64
#define N_BATCH 2

__device__ __forceinline__ float sigmoidf_(float x) {
    return 1.0f / (1.0f + __expf(-x));
}

__device__ __forceinline__ v16bf cat8(v8bf a, v8bf b) {
    return __builtin_shufflevector(a, b, 0,1,2,3,4,5,6,7,8,9,10,11,12,13,14,15);
}

// ---- WMMA fragment loaders (16x16x32 bf16, contiguous k order shared by A and B, wave32) ----
// A: 16x32 (MxK). lane&15 = M; lanes 0-15 hold K {0..15}, lanes 16-31 hold K {16..31}.
__device__ __forceinline__ v16bf load_a_frag(const __bf16* A, int lda, int row0, int kk, int lane) {
    int m  = lane & 15;
    int hi = lane >> 4;
    const __bf16* p = A + (size_t)(row0 + m) * lda + kk + hi * 16;
    v8bf lo = *(const v8bf*)p;         // K = hi*16 + 0..7
    v8bf up = *(const v8bf*)(p + 8);   // K = hi*16 + 8..15
    return cat8(lo, up);
}

// B: 32x16 (KxN) from a K-contiguous (transposed, [N][K]) source: one 32-byte load.
// lane&15 = N; element e <-> K = hi*16 + e.
__device__ __forceinline__ v16bf load_b_frag_kcontig(const __bf16* Bt, int ldk, int kk, int col0, int lane) {
    int n  = lane & 15;
    int hi = lane >> 4;
    return *(const v16bf*)(Bt + (size_t)(col0 + n) * ldk + kk + hi * 16);
}

// ---- f32 -> bf16 conversion ----
__global__ void cvt_f32_bf16_kernel(const float* __restrict__ in, __bf16* __restrict__ out, int n) {
    int i = blockIdx.x * blockDim.x + threadIdx.x;
    if (i < n) out[i] = (__bf16)in[i];
}

// ---- f32 [rows][cols] -> bf16 transposed [cols][rows] ----
__global__ void cvt_transpose_bf16_kernel(const float* __restrict__ in, __bf16* __restrict__ out,
                                          int rows, int cols) {
    int i = blockIdx.x * blockDim.x + threadIdx.x;
    if (i < rows * cols) {
        int r = i / cols, c = i - r * cols;
        out[(size_t)c * rows + r] = (__bf16)in[i];
    }
}

// ---- QKV GEMM: [4096,1024] x Wt[3072,1024] + bias -> head-major Q/K and d-major Vt ----
__global__ void qkv_gemm_kernel(const __bf16* __restrict__ X, const __bf16* __restrict__ Wt,
                                const float* __restrict__ bias,
                                __bf16* __restrict__ Q, __bf16* __restrict__ K, __bf16* __restrict__ Vt) {
    const int lane = threadIdx.x & 31;
    const int gid  = blockIdx.x * (blockDim.x >> 5) + (threadIdx.x >> 5);
    const int tiles_n = (3 * C_EMB) / 64;  // 48
    const int tm = gid / tiles_n;          // 0..255
    const int tn = gid % tiles_n;
    const int row0 = tm * 16, col0 = tn * 64;

    v8f acc[4] = {};
    for (int kk = 0; kk < C_EMB; kk += 32) {
        v16bf a = load_a_frag(X, C_EMB, row0, kk, lane);
#pragma unroll
        for (int t = 0; t < 4; ++t) {
            v16bf b = load_b_frag_kcontig(Wt, C_EMB, kk, col0 + t * 16, lane);
            acc[t] = __builtin_amdgcn_wmma_f32_16x16x32_bf16(false, a, false, b, (short)0, acc[t], false, false);
        }
    }

    const int n = lane & 15, hi = lane >> 4;
#pragma unroll
    for (int t = 0; t < 4; ++t) {
#pragma unroll
        for (int r = 0; r < 8; ++r) {
            int row = row0 + r + 8 * hi;
            int col = col0 + t * 16 + n;
            float val = acc[t][r] + bias[col];
            int region = col / C_EMB;           // 0=q 1=k 2=v
            int cc = col - region * C_EMB;
            int h = cc >> 6, d = cc & 63;
            int bI = row / T_SEQ, tI = row - bI * T_SEQ;
            __bf16 bv = (__bf16)val;
            if (region == 2) {
                // Vt layout: [B,H,D,T] so P*V B-fragments load K-contiguously
                Vt[(((size_t)bI * N_HEAD + h) * D_HEAD + d) * T_SEQ + tI] = bv;
            } else {
                size_t off = (((size_t)bI * N_HEAD + h) * T_SEQ + tI) * D_HEAD + d;
                if (region == 0) Q[off] = bv; else K[off] = bv;
            }
        }
    }
}

// ---- Flash attention with analytic span*wave mask in the numerator ----
__global__ void attn_kernel(const __bf16* __restrict__ Q, const __bf16* __restrict__ K,
                            const __bf16* __restrict__ Vt,
                            const float* __restrict__ span_p, const float* __restrict__ period_w,
                            const float* __restrict__ ratio_w, __bf16* __restrict__ Y) {
    __shared__ __bf16 ldsP[4][16 * 32];
    const int lane = threadIdx.x & 31;
    const int w    = threadIdx.x >> 5;
    const int gid  = blockIdx.x * 4 + w;           // 4096 waves total
    const int tiles_t = T_SEQ / 16;
    const int bh = gid / tiles_t;                   // 0..31
    const int it = gid % tiles_t;
    const int b = bh / N_HEAD, h = bh % N_HEAD;
    const int i0 = it * 16;

    const __bf16* Qh  = Q  + (size_t)bh * T_SEQ * D_HEAD;
    const __bf16* Kh  = K  + (size_t)bh * T_SEQ * D_HEAD;
    const __bf16* Vth = Vt + (size_t)bh * D_HEAD * T_SEQ;

    const float span    = sigmoidf_(span_p[h]) * 2048.0f;
    const float period  = 2.0f + 2.0f * sigmoidf_(period_w[h]);
    const float ratio   = -0.25f + 0.5f * sigmoidf_(ratio_w[h]);
    const float amp     = period * 0.25f;
    const float offs    = period * ratio;
    const float w2pi    = 6.283185307179586f / period;
    const float scale   = 0.125f;  // 1/sqrt(64)
    const float invR    = 1.0f / 32.0f;

    const int n = lane & 15, hi = lane >> 4;

    // Q fragments for the whole D=64 (two K-chunks of 32)
    v16bf qa[2];
#pragma unroll
    for (int c = 0; c < 2; ++c) qa[c] = load_a_frag(Qh, D_HEAD, i0, c * 32, lane);

    v8f o[4] = {};
    float m8[8], l8[8];
#pragma unroll
    for (int r = 0; r < 8; ++r) { m8[r] = -INFINITY; l8[r] = 0.0f; }

    // j0 multiple of 32, j0 <= i0+15 <= 2047 => max key index j0+31 <= 2047 (in bounds)
    for (int j0 = 0; j0 <= i0 + 15; j0 += 32) {
        // S = Q * K^T  (two 16x16 tiles: keys j0..+15 and j0+16..+31)
        v8f s0 = {}, s1 = {};
#pragma unroll
        for (int c = 0; c < 2; ++c) {
            // K^T B-fragment: element e <-> d = c*32 + hi*16 + e, contiguous in memory
            v16bf b0 = *(const v16bf*)(Kh + (size_t)(j0 + n)      * D_HEAD + c * 32 + hi * 16);
            v16bf b1 = *(const v16bf*)(Kh + (size_t)(j0 + 16 + n) * D_HEAD + c * 32 + hi * 16);
            s0 = __builtin_amdgcn_wmma_f32_16x16x32_bf16(false, qa[c], false, b0, (short)0, s0, false, false);
            s1 = __builtin_amdgcn_wmma_f32_16x16x32_bf16(false, qa[c], false, b1, (short)0, s1, false, false);
        }

        float p0v[8], p1v[8];
#pragma unroll
        for (int r = 0; r < 8; ++r) {
            const int i   = i0 + r + 8 * hi;
            const int jj0 = j0 + n;
            const int jj1 = j0 + 16 + n;
            const bool v0 = (jj0 <= i);
            const bool v1 = (jj1 <= i);
            float sm0 = v0 ? s0[r] * scale : -INFINITY;
            float sm1 = v1 ? s1[r] * scale : -INFINITY;
            float mx = fmaxf(sm0, sm1);
#pragma unroll
            for (int off = 1; off < 16; off <<= 1) mx = fmaxf(mx, __shfl_xor(mx, off, 32));
            const float mnew = fmaxf(m8[r], mx);
            const float corr = __expf(m8[r] - mnew);

            float mask0 = 0.0f, mask1 = 0.0f;
            if (v0) {
                float rel = (float)(i - jj0);
                float mp  = fminf(fmaxf((32.0f - rel + span) * invR, 0.0f), 1.0f);
                float wv  = 0.5f * (__cosf(rel * w2pi) + 1.0f) * amp + 0.5f + offs;
                mask0 = mp * fminf(fmaxf(wv, 0.0f), 1.0f);
            }
            if (v1) {
                float rel = (float)(i - jj1);
                float mp  = fminf(fmaxf((32.0f - rel + span) * invR, 0.0f), 1.0f);
                float wv  = 0.5f * (__cosf(rel * w2pi) + 1.0f) * amp + 0.5f + offs;
                mask1 = mp * fminf(fmaxf(wv, 0.0f), 1.0f);
            }
            float p0 = v0 ? __expf(sm0 - mnew) * mask0 : 0.0f;
            float p1 = v1 ? __expf(sm1 - mnew) * mask1 : 0.0f;
            float rs = p0 + p1;
#pragma unroll
            for (int off = 1; off < 16; off <<= 1) rs += __shfl_xor(rs, off, 32);
            l8[r] = l8[r] * corr + rs;
            m8[r] = mnew;
#pragma unroll
            for (int t = 0; t < 4; ++t) o[t][r] *= corr;
            p0v[r] = p0; p1v[r] = p1;
        }

        // Stage P (16x32) in LDS to convert C/D layout -> A layout (same-wave DS ops are in-order)
#pragma unroll
        for (int r = 0; r < 8; ++r) {
            ldsP[w][(r + 8 * hi) * 32 + n]      = (__bf16)p0v[r];
            ldsP[w][(r + 8 * hi) * 32 + 16 + n] = (__bf16)p1v[r];
        }
        asm volatile("s_wait_dscnt 0" ::: "memory");

        v16bf pa;
        {
            const __bf16* p = &ldsP[w][(lane & 15) * 32 + hi * 16];
            pa = cat8(*(const v8bf*)p, *(const v8bf*)(p + 8));
        }
        asm volatile("s_wait_dscnt 0" ::: "memory");

        // O += P * V   (4 n-tiles across D=64); Vt is [D,T] so B-fragments are contiguous
#pragma unroll
        for (int t = 0; t < 4; ++t) {
            v16bf vb = *(const v16bf*)(Vth + (size_t)(t * 16 + n) * T_SEQ + j0 + hi * 16);
            o[t] = __builtin_amdgcn_wmma_f32_16x16x32_bf16(false, pa, false, vb, (short)0, o[t], false, false);
        }
    }

    // Write normalized output, concat-heads layout [B, T, C]
#pragma unroll
    for (int t = 0; t < 4; ++t) {
#pragma unroll
        for (int r = 0; r < 8; ++r) {
            int i = i0 + r + 8 * hi;
            float val = o[t][r] / l8[r];
            size_t off = ((size_t)(b * T_SEQ + i)) * C_EMB + h * D_HEAD + t * 16 + n;
            Y[off] = (__bf16)val;
        }
    }
}

// ---- Output projection: [4096,1024] x Wt[1024,1024] + bias -> f32 ----
__global__ void proj_gemm_kernel(const __bf16* __restrict__ Y, const __bf16* __restrict__ Wt,
                                 const float* __restrict__ bias, float* __restrict__ out) {
    const int lane = threadIdx.x & 31;
    const int gid  = blockIdx.x * (blockDim.x >> 5) + (threadIdx.x >> 5);
    const int tiles_n = C_EMB / 64;   // 16
    const int tm = gid / tiles_n;
    const int tn = gid % tiles_n;
    const int row0 = tm * 16, col0 = tn * 64;

    v8f acc[4] = {};
    for (int kk = 0; kk < C_EMB; kk += 32) {
        v16bf a = load_a_frag(Y, C_EMB, row0, kk, lane);
#pragma unroll
        for (int t = 0; t < 4; ++t) {
            v16bf b = load_b_frag_kcontig(Wt, C_EMB, kk, col0 + t * 16, lane);
            acc[t] = __builtin_amdgcn_wmma_f32_16x16x32_bf16(false, a, false, b, (short)0, acc[t], false, false);
        }
    }

    const int n = lane & 15, hi = lane >> 4;
#pragma unroll
    for (int t = 0; t < 4; ++t) {
#pragma unroll
        for (int r = 0; r < 8; ++r) {
            int row = row0 + r + 8 * hi;
            int col = col0 + t * 16 + n;
            out[(size_t)row * C_EMB + col] = acc[t][r] + bias[col];
        }
    }
}

// ---- Scalar span loss ----
__global__ void span_loss_kernel(const float* __restrict__ sp, const float* __restrict__ pw,
                                 const float* __restrict__ rw, float* __restrict__ out, int idx) {
    if (threadIdx.x == 0 && blockIdx.x == 0) {
        float acc = 0.0f;
        for (int h = 0; h < N_HEAD; ++h) {
            float span   = sigmoidf_(sp[h]) * 2048.0f;
            float period = 2.0f + 2.0f * sigmoidf_(pw[h]);
            float ratio  = -0.25f + 0.5f * sigmoidf_(rw[h]);
            float lt = 1.0f / period + 2.0f * ratio - 0.25f + 0.5f;
            acc += (span + 32.0f) * lt;
        }
        out[idx] = 2e-6f * acc / (float)N_HEAD;
    }
}

extern "C" void kernel_launch(void* const* d_in, const int* in_sizes, int n_in,
                              void* d_out, int out_size, void* d_ws, size_t ws_size,
                              hipStream_t stream) {
    const float* x       = (const float*)d_in[0];
    const float* w_attn  = (const float*)d_in[1];
    const float* b_attn  = (const float*)d_in[2];
    const float* w_proj  = (const float*)d_in[3];
    const float* b_proj  = (const float*)d_in[4];
    const float* span_p  = (const float*)d_in[5];
    const float* per_w   = (const float*)d_in[6];
    const float* rat_w   = (const float*)d_in[7];
    float* out = (float*)d_out;

    char* ws = (char*)d_ws;
    const size_t nX  = (size_t)N_BATCH * T_SEQ * C_EMB;       // 4194304
    const size_t nWA = (size_t)C_EMB * 3 * C_EMB;             // 3145728
    const size_t nWP = (size_t)C_EMB * C_EMB;                 // 1048576
    __bf16* xbf  = (__bf16*)(ws);
    __bf16* wat  = (__bf16*)(ws + nX * 2);                    // transposed [3072][1024]
    __bf16* wpt  = (__bf16*)(ws + nX * 2 + nWA * 2);          // transposed [1024][1024]
    __bf16* Qbf  = (__bf16*)(ws + nX * 2 + nWA * 2 + nWP * 2);
    __bf16* Kbf  = Qbf + nX;
    __bf16* Vtbf = Kbf + nX;                                  // [B,H,D,T]
    __bf16* Ybf  = Vtbf + nX;

    cvt_f32_bf16_kernel<<<(int)((nX + 255) / 256), 256, 0, stream>>>(x, xbf, (int)nX);
    cvt_transpose_bf16_kernel<<<(int)((nWA + 255) / 256), 256, 0, stream>>>(w_attn, wat, C_EMB, 3 * C_EMB);
    cvt_transpose_bf16_kernel<<<(int)((nWP + 255) / 256), 256, 0, stream>>>(w_proj, wpt, C_EMB, C_EMB);

    // 256 M-tiles * 48 N-strips = 12288 waves, 8 waves/block
    qkv_gemm_kernel<<<1536, 256, 0, stream>>>(xbf, wat, b_attn, Qbf, Kbf, Vtbf);

    // B*H*(T/16) = 4096 waves, 4 waves/block
    attn_kernel<<<1024, 128, 0, stream>>>(Qbf, Kbf, Vtbf, span_p, per_w, rat_w, Ybf);

    // 256 * 16 = 4096 waves, 8 waves/block
    proj_gemm_kernel<<<512, 256, 0, stream>>>(Ybf, wpt, b_proj, out);

    span_loss_kernel<<<1, 32, 0, stream>>>(span_p, per_w, rat_w, out, out_size - 1);
}
